// BiMambaBlock_2130303779106
// MI455X (gfx1250) — matrix operand order repaired and checked by reading, not compiled, not run on hardware
//
#include <hip/hip_runtime.h>
#include <math.h>

// ---------------- problem constants ----------------
#define D_MODEL 1024
#define D_STATE 16
#define D_CONV  4
#define D_INNER 2048
#define DT_RANK 64
#define BATCH   4
#define SEQ     1024
#define NROWS   (BATCH * SEQ)          // 4096 token rows
#define XZ_W    (2 * D_INNER)          // 4096
#define DBL_W   (DT_RANK + 2*D_STATE)  // 96

// ---------------- CDNA5 WMMA types ----------------
typedef __attribute__((ext_vector_type(16))) __bf16 v16bf;
typedef __attribute__((ext_vector_type(8)))  __bf16 v8bf;
typedef __attribute__((ext_vector_type(8)))  float  v8f;

__device__ __forceinline__ float sigmoidf_(float x) { return 1.0f / (1.0f + __expf(-x)); }
__device__ __forceinline__ float siluf_(float x)    { return x * sigmoidf_(x); }
__device__ __forceinline__ float softplusf_(float x){ return (x > 20.0f) ? x : log1pf(__expf(x)); }

// Async copy: 16 bytes global -> LDS per active lane (gfx1250 ASYNCcnt path)
__device__ __forceinline__ void async_copy_b128(uint32_t lds_off, const void* gptr) {
    asm volatile("global_load_async_to_lds_b128 %0, %1, off"
                 :: "v"(lds_off), "v"(gptr) : "memory");
}
__device__ __forceinline__ void wait_async0() {
    asm volatile("s_wait_asynccnt 0x0" ::: "memory");
}
__device__ __forceinline__ uint32_t lds_off32(const void* p) {
    // generic pointers to LDS carry the LDS offset in the low 32 bits
    return (uint32_t)(uintptr_t)p;
}

// =====================================================================
// GEMM:  C[M,N] (+)= act( A[M,K] @ W[K,N] + bias ),  W given TRANSPOSED
// as Wt[N][K].  A and Wt are bf16 in memory.  Tiles stream to LDS via
// GLOBAL_LOAD_ASYNC_TO_LDS_B128 with double buffering; math is
// v_wmma_f32_16x16x32_bf16 with f32 accumulation.
// BLOCK 128x128, BK=32, 256 threads = 8 waves (4Mx2N); wave = 2x4 tiles.
// Requires M % 128 == 0, K % 32 == 0.  GUARD=true: N-bounds (N=96 GEMM).
// =====================================================================
template<bool GUARD>
__global__ __launch_bounds__(256)
void gemm_bf16_wmma(const __bf16* __restrict__ A, int lda,   // [M,K]
                    const __bf16* __restrict__ Wt, int ldw,  // [N,K]
                    float* __restrict__ C, int ldc,
                    __bf16* __restrict__ Cbf,                // optional bf16 mirror
                    const float* __restrict__ bias,
                    int N, int K, int accum, int act)
{
    __shared__ __attribute__((aligned(16))) __bf16 As[2][128][40]; // [m][k], 80B rows
    __shared__ __attribute__((aligned(16))) __bf16 Bs[2][128][40]; // [n][k]

    const int tid  = threadIdx.x;
    const int m0   = blockIdx.y * 128;
    const int n0   = blockIdx.x * 128;
    const int wave = tid >> 5;
    const int lane = tid & 31;
    const int wm   = (wave & 3) * 32;   // 4 waves in M
    const int wn   = (wave >> 2) * 64;  // 2 waves in N
    const int hl   = lane >> 4;
    const int lr   = lane & 15;

    v8f acc[2][4];
    #pragma unroll
    for (int i = 0; i < 2; ++i)
        #pragma unroll
        for (int j = 0; j < 4; ++j) { v8f z = {}; acc[i][j] = z; }

    // Each tile (128 rows x 32 bf16 = 8KB) = 512 x 16B chunks, 2 per thread.
    auto issue_tile = [&](int k0, int buf) {
        #pragma unroll
        for (int i = 0; i < 2; ++i) {
            int idx = tid + i * 256;          // 0..511
            int r   = idx >> 2;               // 4 chunks per 32-wide row
            int c   = (idx & 3) << 3;         // bf16 col: 0,8,16,24
            async_copy_b128(lds_off32(&As[buf][r][c]),
                            A + (size_t)(m0 + r) * lda + (k0 + c));
        }
        #pragma unroll
        for (int i = 0; i < 2; ++i) {
            int idx = tid + i * 256;
            int n   = idx >> 2;
            int c   = (idx & 3) << 3;
            if (!GUARD || (n0 + n) < N)       // stale cols >= N are never stored
                async_copy_b128(lds_off32(&Bs[buf][n][c]),
                                Wt + (size_t)(n0 + n) * ldw + (k0 + c));
        }
    };

    const int nk = K >> 5;
    issue_tile(0, 0);
    for (int kt = 0; kt < nk; ++kt) {
        const int buf = kt & 1;
        wait_async0();                         // my wave's copies done
        __syncthreads();                       // everyone's copies visible
        if (kt + 1 < nk)
            issue_tile((kt + 1) << 5, buf ^ 1); // overlaps WMMA below

        // ---- fragments per documented CDNA5 16-bit VGPR layouts ----
        v16bf afrag[2], bfrag[4];
        #pragma unroll
        for (int i = 0; i < 2; ++i) {
            int mrow = wm + i * 16 + lr;
            v8bf lo = *(const v8bf*)&As[buf][mrow][hl * 8];
            v8bf hi = *(const v8bf*)&As[buf][mrow][16 + hl * 8];
            afrag[i] = __builtin_shufflevector(lo, hi, 0,1,2,3,4,5,6,7,8,9,10,11,12,13,14,15);
        }
        #pragma unroll
        for (int j = 0; j < 4; ++j) {
            int ncol = wn + j * 16 + lr;
            v8bf lo = *(const v8bf*)&Bs[buf][ncol][hl * 8];
            v8bf hi = *(const v8bf*)&Bs[buf][ncol][16 + hl * 8];
            bfrag[j] = __builtin_shufflevector(lo, hi, 0,1,2,3,4,5,6,7,8,9,10,11,12,13,14,15);
        }
        #pragma unroll
        for (int i = 0; i < 2; ++i)
            #pragma unroll
            for (int j = 0; j < 4; ++j)
                acc[i][j] = __builtin_amdgcn_wmma_f32_16x16x32_bf16(
                    false, afrag[i], false, bfrag[j], (short)0, acc[i][j], false, false);
    }

    // ---- epilogue: VGPR r -> M = base + hl*8 + r, N = base + lr ----
    #pragma unroll
    for (int i = 0; i < 2; ++i) {
        #pragma unroll
        for (int j = 0; j < 4; ++j) {
            int col = n0 + wn + j * 16 + lr;
            if (GUARD && col >= N) continue;
            int rowbase = m0 + wm + i * 16 + hl * 8;
            #pragma unroll
            for (int r = 0; r < 8; ++r) {
                size_t off = (size_t)(rowbase + r) * ldc + col;
                float v = acc[i][j][r];
                if (accum) v += C[off];
                if (bias)  v += bias[col];
                if (act == 1) v = softplusf_(v);
                C[off] = v;
                if (Cbf) Cbf[off] = (__bf16)v;
            }
        }
    }
}

// =====================================================================
// Weight prep: f32 [K,N] -> bf16 [N,K] (tiled transpose+convert)
// =====================================================================
__global__ __launch_bounds__(256)
void transpose_cvt_kernel(const float* __restrict__ in, __bf16* __restrict__ out,
                          int K, int N)
{
    __shared__ float tile[32][33];
    int kb = blockIdx.x * 32, nb = blockIdx.y * 32;
    int tx = threadIdx.x & 31, ty = threadIdx.x >> 5;   // 32 x 8
    #pragma unroll
    for (int i = 0; i < 32; i += 8) {
        int k = kb + ty + i, n = nb + tx;
        tile[ty + i][tx] = (k < K && n < N) ? in[(size_t)k * N + n] : 0.0f;
    }
    __syncthreads();
    #pragma unroll
    for (int i = 0; i < 32; i += 8) {
        int n = nb + ty + i, k = kb + tx;
        if (n < N && k < K) out[(size_t)n * K + k] = (__bf16)tile[tx][ty + i];
    }
}

// flat f32 -> bf16
__global__ void cvt_bf16_kernel(const float* __restrict__ in, __bf16* __restrict__ out,
                                size_t total)
{
    size_t idx = (size_t)blockIdx.x * blockDim.x + threadIdx.x;
    if (idx < total) out[idx] = (__bf16)in[idx];
}

// out_bf[b,t,:] = bf16(in[b,L-1-t,:])
__global__ void flip_cvt_kernel(const float* __restrict__ in, __bf16* __restrict__ out,
                                int Wd)
{
    size_t idx = (size_t)blockIdx.x * blockDim.x + threadIdx.x;
    size_t total = (size_t)NROWS * Wd;
    if (idx >= total) return;
    int c = (int)(idx % Wd);
    size_t bt = idx / Wd;
    int t = (int)(bt % SEQ);
    size_t brow = bt - t;
    out[idx] = (__bf16)in[(brow + (SEQ - 1 - t)) * (size_t)Wd + c];
}

// flip producing both f32 and bf16
__global__ void flip_dual_kernel(const float* __restrict__ in,
                                 float* __restrict__ outf, __bf16* __restrict__ outb,
                                 int Wd)
{
    size_t idx = (size_t)blockIdx.x * blockDim.x + threadIdx.x;
    size_t total = (size_t)NROWS * Wd;
    if (idx >= total) return;
    int c = (int)(idx % Wd);
    size_t bt = idx / Wd;
    int t = (int)(bt % SEQ);
    size_t brow = bt - t;
    float v = in[(brow + (SEQ - 1 - t)) * (size_t)Wd + c];
    outf[idx] = v;
    outb[idx] = (__bf16)v;
}

// =====================================================================
// Depthwise causal conv (width 4) + SiLU; emits f32 (scan) + bf16 (GEMM)
// =====================================================================
__global__ void conv_silu_kernel(const float* __restrict__ xz,
                                 const float* __restrict__ cw,
                                 const float* __restrict__ cb,
                                 float* __restrict__ uc,
                                 __bf16* __restrict__ ucb)
{
    size_t idx = (size_t)blockIdx.x * blockDim.x + threadIdx.x;
    size_t total = (size_t)NROWS * D_INNER;
    if (idx >= total) return;
    int d = (int)(idx % D_INNER);
    size_t bt = idx / D_INNER;
    int t = (int)(bt % SEQ);
    size_t brow = bt - t;
    float acc = cb[d];
    #pragma unroll
    for (int k = 0; k < D_CONV; ++k) {
        int tt = t + k - (D_CONV - 1);
        if (tt >= 0)
            acc += xz[(brow + tt) * (size_t)XZ_W + d] * cw[d * D_CONV + k];
    }
    float v = siluf_(acc);
    uc[idx]  = v;
    ucb[idx] = (__bf16)v;
}

// =====================================================================
// Selective scan: one thread per (batch, channel); h[16] in registers.
// =====================================================================
__global__ void scan_kernel(const float* __restrict__ uc,
                            const float* __restrict__ delta,
                            const float* __restrict__ dbl,
                            const float* __restrict__ Alog,
                            float* __restrict__ ys)
{
    int idx = blockIdx.x * blockDim.x + threadIdx.x;
    if (idx >= BATCH * D_INNER) return;
    int d = idx % D_INNER;
    int b = idx / D_INNER;

    float Ar[D_STATE];
    #pragma unroll
    for (int n = 0; n < D_STATE; ++n) Ar[n] = -__expf(Alog[d * D_STATE + n]);
    float h[D_STATE];
    #pragma unroll
    for (int n = 0; n < D_STATE; ++n) h[n] = 0.0f;

    for (int t = 0; t < SEQ; ++t) {
        size_t row = (size_t)b * SEQ + t;
        float dlt = delta[row * D_INNER + d];
        float ut  = uc[row * D_INNER + d];
        const float* bc = dbl + row * DBL_W;    // [dt(64) | B(16) | C(16)]
        float du = dlt * ut;
        float y = 0.0f;
        #pragma unroll
        for (int n = 0; n < D_STATE; ++n) {
            float dA = __expf(dlt * Ar[n]);
            h[n] = dA * h[n] + du * bc[DT_RANK + n];
            y += h[n] * bc[DT_RANK + D_STATE + n];
        }
        ys[row * D_INNER + d] = y;
    }
}

// =====================================================================
// ycomb = bf16( (ys + uc * Dp) * silu(z) ),  z = xz[:, :, Di:2Di]
// =====================================================================
__global__ void combine_kernel(const float* __restrict__ ys,
                               const float* __restrict__ uc,
                               const float* __restrict__ Dp,
                               const float* __restrict__ xz,
                               __bf16* __restrict__ yout)
{
    size_t idx = (size_t)blockIdx.x * blockDim.x + threadIdx.x;
    size_t total = (size_t)NROWS * D_INNER;
    if (idx >= total) return;
    int d = (int)(idx % D_INNER);
    size_t row = idx / D_INNER;
    float z = xz[row * (size_t)XZ_W + D_INNER + d];
    yout[idx] = (__bf16)((ys[idx] + uc[idx] * Dp[d]) * siluf_(z));
}

// =====================================================================
// g = sigmoid(gpre);  yg = bf16( g*yf + (1-g)*yb )
// =====================================================================
__global__ void blend_kernel(const float* __restrict__ gpre,
                             const float* __restrict__ yf,
                             const float* __restrict__ yb,
                             __bf16* __restrict__ yg)
{
    size_t idx = (size_t)blockIdx.x * blockDim.x + threadIdx.x;
    size_t total = (size_t)NROWS * D_MODEL;
    if (idx >= total) return;
    float g = sigmoidf_(gpre[idx]);
    yg[idx] = (__bf16)(g * yf[idx] + (1.0f - g) * yb[idx]);
}

// =====================================================================
// Host side
// =====================================================================
extern "C" void kernel_launch(void* const* d_in, const int* in_sizes, int n_in,
                              void* d_out, int out_size, void* d_ws, size_t ws_size,
                              hipStream_t stream)
{
    const float* x = (const float*)d_in[0];
    const float* gate_w = (const float*)d_in[19];
    const float* gate_b = (const float*)d_in[20];
    const float* proj_w = (const float*)d_in[21];
    const float* proj_b = (const float*)d_in[22];
    float* out = (float*)d_out;
    (void)ws_size; (void)n_in; (void)in_sizes; (void)out_size;

    char* ws = (char*)d_ws;
    size_t off = 0;
    auto carveB = [&](size_t bytes) -> char* {
        char* p = ws + off;
        off += (bytes + 255) & ~(size_t)255;
        return p;
    };
    auto carveF  = [&](size_t elems) { return (float*)carveB(elems * 4); };
    auto carveH  = [&](size_t elems) { return (__bf16*)carveB(elems * 2); };

    // f32 buffers
    float* buf_xz    = carveF((size_t)NROWS * XZ_W);
    float* buf_uc    = carveF((size_t)NROWS * D_INNER);
    float* buf_dbl   = carveF((size_t)NROWS * DBL_W);
    float* buf_delta = carveF((size_t)NROWS * D_INNER);
    float* buf_ys    = carveF((size_t)NROWS * D_INNER);
    float* buf_yf    = carveF((size_t)NROWS * D_MODEL);
    float* buf_yb    = carveF((size_t)NROWS * D_MODEL);
    float* buf_ybt   = carveF((size_t)NROWS * D_MODEL);
    float* buf_gpre  = carveF((size_t)NROWS * D_MODEL);
    // bf16 activation buffers
    __bf16* x_bf     = carveH((size_t)NROWS * D_MODEL);
    __bf16* xrev_bf  = carveH((size_t)NROWS * D_MODEL);
    __bf16* uc_bf    = carveH((size_t)NROWS * D_INNER);
    __bf16* dbl_bf   = carveH((size_t)NROWS * DBL_W);
    __bf16* ycomb_bf = carveH((size_t)NROWS * D_INNER);
    __bf16* yf_bf    = carveH((size_t)NROWS * D_MODEL);
    __bf16* yb_bf    = carveH((size_t)NROWS * D_MODEL);
    __bf16* yg_bf    = carveH((size_t)NROWS * D_MODEL);
    // bf16 transposed weights [N][K]
    __bf16* inprojT[2] = { carveH((size_t)D_MODEL * XZ_W), carveH((size_t)D_MODEL * XZ_W) };
    __bf16* xprojT[2]  = { carveH((size_t)D_INNER * DBL_W), carveH((size_t)D_INNER * DBL_W) };
    __bf16* dtwT[2]    = { carveH((size_t)DT_RANK * D_INNER), carveH((size_t)DT_RANK * D_INNER) };
    __bf16* outprojT[2]= { carveH((size_t)D_INNER * D_MODEL), carveH((size_t)D_INNER * D_MODEL) };
    __bf16* gatewT     = carveH((size_t)(2*D_MODEL) * D_MODEL);
    __bf16* projwT     = carveH((size_t)D_MODEL * D_MODEL);

    auto transpose = [&](const float* w, __bf16* wt, int K, int N) {
        dim3 grid((K + 31) / 32, (N + 31) / 32);
        transpose_cvt_kernel<<<grid, 256, 0, stream>>>(w, wt, K, N);
    };
    auto gemm = [&](const __bf16* A, int lda, const __bf16* Wt, int ldw,
                    float* C, int ldc, __bf16* Cbf, const float* bias,
                    int M, int N, int K, int accum, int act) {
        dim3 grid((N + 127) / 128, M / 128);
        if (N % 128 == 0)
            gemm_bf16_wmma<false><<<grid, 256, 0, stream>>>(A, lda, Wt, ldw, C, ldc, Cbf, bias, N, K, accum, act);
        else
            gemm_bf16_wmma<true><<<grid, 256, 0, stream>>>(A, lda, Wt, ldw, C, ldc, Cbf, bias, N, K, accum, act);
    };
    auto ew = [&](size_t total) { return dim3((unsigned)((total + 255) / 256)); };

    // ---------- weight prep (bf16 transposed) ----------
    for (int s = 0; s < 2; ++s) {
        int pb = 1 + s * 9;
        transpose((const float*)d_in[pb + 0], inprojT[s],  D_MODEL, XZ_W);
        transpose((const float*)d_in[pb + 3], xprojT[s],   D_INNER, DBL_W);
        transpose((const float*)d_in[pb + 4], dtwT[s],     DT_RANK, D_INNER);
        transpose((const float*)d_in[pb + 8], outprojT[s], D_INNER, D_MODEL);
    }
    transpose(gate_w, gatewT, 2 * D_MODEL, D_MODEL);
    transpose(proj_w, projwT, D_MODEL, D_MODEL);
    cvt_bf16_kernel<<<ew((size_t)NROWS * D_MODEL), 256, 0, stream>>>(
        x, x_bf, (size_t)NROWS * D_MODEL);
    flip_cvt_kernel<<<ew((size_t)NROWS * D_MODEL), 256, 0, stream>>>(x, xrev_bf, D_MODEL);

    auto run_dir = [&](const __bf16* Xbf, int pb, const __bf16* inpT, const __bf16* xpT,
                       const __bf16* dtT, const __bf16* opT,
                       float* Ydir, __bf16* Ydir_bf) {
        const float* conv_w  = (const float*)d_in[pb + 1];
        const float* conv_b  = (const float*)d_in[pb + 2];
        const float* dt_b    = (const float*)d_in[pb + 5];
        const float* Alog    = (const float*)d_in[pb + 6];
        const float* Dp      = (const float*)d_in[pb + 7];

        // xz = X @ inproj
        gemm(Xbf, D_MODEL, inpT, D_MODEL, buf_xz, XZ_W, nullptr, nullptr,
             NROWS, XZ_W, D_MODEL, 0, 0);
        // uc = silu(causal_conv(u))
        conv_silu_kernel<<<ew((size_t)NROWS * D_INNER), 256, 0, stream>>>(
            buf_xz, conv_w, conv_b, buf_uc, uc_bf);
        // dbl = uc @ xproj   (N=96, GUARD path)
        gemm(uc_bf, D_INNER, xpT, D_INNER, buf_dbl, DBL_W, dbl_bf, nullptr,
             NROWS, DBL_W, D_INNER, 0, 0);
        // delta = softplus(dt @ dt_w + dt_b)
        gemm(dbl_bf, DBL_W, dtT, DT_RANK, buf_delta, D_INNER, nullptr, dt_b,
             NROWS, D_INNER, DT_RANK, 0, /*softplus*/1);
        // selective scan
        scan_kernel<<<(BATCH * D_INNER + 255) / 256, 256, 0, stream>>>(
            buf_uc, buf_delta, buf_dbl, Alog, buf_ys);
        // ycomb = (ys + uc*Dp) * silu(z)  -> bf16
        combine_kernel<<<ew((size_t)NROWS * D_INNER), 256, 0, stream>>>(
            buf_ys, buf_uc, Dp, buf_xz, ycomb_bf);
        // Ydir = ycomb @ outproj
        gemm(ycomb_bf, D_INNER, opT, D_INNER, Ydir, D_MODEL, Ydir_bf, nullptr,
             NROWS, D_MODEL, D_INNER, 0, 0);
    };

    // ---------- forward ----------
    run_dir(x_bf, 1, inprojT[0], xprojT[0], dtwT[0], outprojT[0], buf_yf, yf_bf);
    // ---------- backward (flip, run, flip back) ----------
    run_dir(xrev_bf, 10, inprojT[1], xprojT[1], dtwT[1], outprojT[1], buf_ybt, nullptr);
    flip_dual_kernel<<<ew((size_t)NROWS * D_MODEL), 256, 0, stream>>>(
        buf_ybt, buf_yb, yb_bf, D_MODEL);

    // ---------- gate: g = sigmoid([yf|yb] @ gate_w + gate_b) ----------
    gemm(yf_bf, D_MODEL, gatewT, 2 * D_MODEL, buf_gpre, D_MODEL, nullptr, nullptr,
         NROWS, D_MODEL, D_MODEL, 0, 0);
    gemm(yb_bf, D_MODEL, gatewT + D_MODEL, 2 * D_MODEL, buf_gpre, D_MODEL, nullptr, gate_b,
         NROWS, D_MODEL, D_MODEL, /*accum*/1, 0);
    blend_kernel<<<ew((size_t)NROWS * D_MODEL), 256, 0, stream>>>(
        buf_gpre, buf_yf, buf_yb, yg_bf);

    // ---------- final projection ----------
    gemm(yg_bf, D_MODEL, projwT, D_MODEL, out, D_MODEL, nullptr, proj_b,
         NROWS, D_MODEL, D_MODEL, 0, 0);
}
